// Encoder_88261577932984
// MI455X (gfx1250) — matrix operand order repaired and checked by reading, not compiled, not run on hardware
//
#include <hip/hip_runtime.h>
#include <hip/hip_bf16.h>

// Problem constants (from the reference)
#define BB   512          // batch
#define TM1  255          // time steps (T-1)
#define KK   256          // input size (driving series)
#define HH   256          // hidden size
#define SP   256          // padded s-dimension (attention length, 255 -> 256)
#define OUT_ENC_OFF (BB * TM1 * KK)   // d_out: [input_weighted | input_encoded]

typedef unsigned short u16;
typedef __attribute__((ext_vector_type(8)))  u16    v8u;
typedef __attribute__((ext_vector_type(16))) u16    v16u;
typedef __attribute__((ext_vector_type(16))) __bf16 v16bf;
typedef __attribute__((ext_vector_type(8)))  float  v8f;

#define LOG2E 1.4426950408889634f

// ---------- bf16 helpers: use native hardware converts via __bf16 ----------
__device__ __forceinline__ u16 f2bf(float f) {
    __bf16 b = (__bf16)f;                 // RNE, lowers to v_cvt on gfx1250
    return __builtin_bit_cast(u16, b);
}
__device__ __forceinline__ float bf2f(u16 h) {
    return (float)__builtin_bit_cast(__bf16, h);
}

// ---------- fast transcendental helpers (native TRANS ops) ----------
__device__ __forceinline__ float fast_exp(float x) {
    return __builtin_amdgcn_exp2f(x * LOG2E);
}
__device__ __forceinline__ float fast_tanh(float x) {
#if __has_builtin(__builtin_amdgcn_tanhf)
    return __builtin_amdgcn_tanhf(x);          // V_TANH_F32
#else
    float t = __builtin_amdgcn_exp2f(x * (2.0f * LOG2E));   // e^{2x}
    return (t - 1.0f) * __builtin_amdgcn_rcpf(t + 1.0f);
#endif
}
__device__ __forceinline__ float fast_sigm(float x) {
    return __builtin_amdgcn_rcpf(1.0f + __builtin_amdgcn_exp2f(-LOG2E * x));
}

// ---------- WMMA wrapper: D = A(16x32 bf16) * B(32x16 bf16) + C(f32) ----------
__device__ __forceinline__ v8f wmma_bf16(v16u a, v16u b, v8f c) {
    return __builtin_amdgcn_wmma_f32_16x16x32_bf16(
        false, __builtin_bit_cast(v16bf, a),
        false, __builtin_bit_cast(v16bf, b),
        (short)0, c, false, false);
}
// two 16-byte spans -> one 16-half fragment register set
__device__ __forceinline__ v16u ld16(const u16* p0, const u16* p1) {
    v8u lo = *(const v8u*)p0;
    v8u hi = *(const v8u*)p1;
    v16u r;
#pragma unroll
    for (int i = 0; i < 8; ++i) { r[i] = lo[i]; r[i + 8] = hi[i]; }
    return r;
}

// =========================== setup kernels ===========================
__global__ void k_init(float* __restrict__ c_state, u16* __restrict__ hc0) {
    int i = blockIdx.x * blockDim.x + threadIdx.x;   // 0 .. B*H-1
    c_state[i] = 0.0f;
    hc0[i] = 0;                 // first half (rows 0..255, h and c)
    hc0[i + BB * HH] = 0;       // second half (rows 256..511, h and c)
}

__global__ void k_cvt(u16* __restrict__ dst, const float* __restrict__ src, int n) {
    int i = blockIdx.x * blockDim.x + threadIdx.x;
    if (i < n) dst[i] = f2bf(src[i]);
}

// U_e padded to [256 s][256 t] bf16 (zero row/col 255)
__global__ void k_build_ue(u16* __restrict__ dst, const float* __restrict__ Ue) {
    int i = blockIdx.x * blockDim.x + threadIdx.x;   // 65536
    int s = i >> 8, t = i & 255;
    float v = (s < TM1 && t < TM1) ? Ue[s * TM1 + t] : 0.0f;
    dst[i] = f2bf(v);
}

// bias2[s] = W_ehs_b[s] + U_e_b[s] (folded; softmax makes v_e_b irrelevant)
// bsum[j]  = b_ih[j] + b_hh[j]
__global__ void k_build_bias(float* __restrict__ bias2, float* __restrict__ bsum,
                             const float* __restrict__ Wb, const float* __restrict__ Ub,
                             const float* __restrict__ bih, const float* __restrict__ bhh) {
    int i = blockIdx.x * blockDim.x + threadIdx.x;   // 1024
    bsum[i] = bih[i] + bhh[i];
    if (i < SP) bias2[i] = (i < TM1) ? (Wb[i] + Ub[i]) : 0.0f;
}

// =========================== u pre-GEMM (WMMA) ===========================
// u_t[b][s][k] = sum_t U_e[s,t] * input[b,t,k]   (bf16 output, L2-resident 64MB)
// A = U_e (16x32 tiles), B = input[b] slab ([t][k], k contiguous -> coalesced)
// Block = (b, ktile); 4 waves x 4 s-tiles; B fragments reused across s-tiles
// so `input` is read from HBM exactly once.
__global__ void __launch_bounds__(128) k_ugemm(u16* __restrict__ u_t,
                                               const u16* __restrict__ Uebf,
                                               const float* __restrict__ input) {
    const int b     = blockIdx.x >> 4;
    const int ktile = blockIdx.x & 15;
    const int widx  = threadIdx.x >> 5;
    const int lane  = threadIdx.x & 31;
    const int lhalf = (lane & 16) ? 1 : 0;
    const int l15   = lane & 15;
    const int kcol  = ktile * 16 + l15;
    const float* inb = input + (size_t)b * (TM1 * KK) + kcol;

    v8f z = {0.f, 0.f, 0.f, 0.f, 0.f, 0.f, 0.f, 0.f};
    v8f acc[4];
#pragma unroll
    for (int si = 0; si < 4; ++si) acc[si] = z;

    for (int c = 0; c < 8; ++c) {
        const int ct = c * 32;
        // B fragment: K = t rows (guard t==255 pad), N = k column
        const int tbase = ct + lhalf * 8;
        v16u bfr;
#pragma unroll
        for (int j = 0; j < 16; ++j) {
            int t = tbase + (j < 8 ? j : j + 8);
            float v = (t < TM1) ? inb[t * KK] : 0.0f;
            bfr[j] = f2bf(v);
        }
#pragma unroll
        for (int si = 0; si < 4; ++si) {
            const int stile = widx * 4 + si;
            const u16* arow = Uebf + (stile * 16 + l15) * SP + ct + lhalf * 8;
            v16u afr = ld16(arow, arow + 16);
            acc[si] = wmma_bf16(afr, bfr, acc[si]);
        }
    }
#pragma unroll
    for (int si = 0; si < 4; ++si) {
        const int stile = widx * 4 + si;
#pragma unroll
        for (int r = 0; r < 8; ++r) {
            int s = stile * 16 + r + lhalf * 8;
            u_t[((size_t)b * SP + s) * KK + kcol] = f2bf(acc[si][r]);
        }
    }
}

// =========================== per-step attention ===========================
// Block b: (1) w[s] = [h,c]·W_ehs[s,:] + (W_ehs_b+U_e_b)[s]  (VALU dot, LDS)
//          (2) v[k] = sum_s tanh(w[s]+u[b,s,k])·v_w[s]  (u coalesced from L2)
//          (3) softmax over k, x_hat = a*x_t -> d_out + bf16 copy for LSTM GEMM
__global__ void __launch_bounds__(256) k_attn(const u16* __restrict__ u_t,
                                              const u16* __restrict__ hc_in,
                                              const u16* __restrict__ Wehsbf,
                                              const float* __restrict__ bias2,
                                              const float* __restrict__ vew,
                                              const float* __restrict__ input,
                                              float* __restrict__ out_w,
                                              u16* __restrict__ xhbf, int t) {
    __shared__ float hc_s[2 * HH];
    __shared__ float w_s[SP];
    __shared__ float vw_s[SP];
    __shared__ float red[KK];
    const int b = blockIdx.x;
    const int tid = threadIdx.x;

    hc_s[tid]       = bf2f(hc_in[b * (2 * HH) + tid]);
    hc_s[tid + 256] = bf2f(hc_in[b * (2 * HH) + 256 + tid]);
    vw_s[tid] = (tid < TM1) ? vew[tid] : 0.0f;
    __syncthreads();

    {   // phase A: w[s]
        float acc = 0.0f;
        if (tid < TM1) {
            const u16* row = Wehsbf + tid * (2 * HH);
            for (int j = 0; j < 2 * HH; j += 8) {
                v8u x = *(const v8u*)(row + j);
#pragma unroll
                for (int q = 0; q < 8; ++q) acc += hc_s[j + q] * bf2f(x[q]);
            }
            acc += bias2[tid];
        }
        w_s[tid] = acc;
    }
    __syncthreads();

    // phase B: score for k = tid  (u[b][s][k]: fixed s -> contiguous over k)
    const u16* up = u_t + (size_t)b * (SP * KK) + tid;
    float v = 0.0f;
#pragma unroll 4
    for (int s = 0; s < TM1; ++s)
        v += fast_tanh(w_s[s] + bf2f(up[s * KK])) * vw_s[s];

    // softmax over K=256 (v_e_b dropped: shift-invariant)
    red[tid] = v; __syncthreads();
    for (int off = 128; off > 0; off >>= 1) {
        if (tid < off) red[tid] = fmaxf(red[tid], red[tid + off]);
        __syncthreads();
    }
    float m = red[0]; __syncthreads();
    float e = fast_exp(v - m);
    red[tid] = e; __syncthreads();
    for (int off = 128; off > 0; off >>= 1) {
        if (tid < off) red[tid] += red[tid + off];
        __syncthreads();
    }
    float a = e * __builtin_amdgcn_rcpf(red[0]);

    float xh = a * input[(size_t)b * (TM1 * KK) + t * KK + tid];
    out_w[(size_t)b * (TM1 * KK) + t * KK + tid] = xh;
    xhbf[b * KK + tid] = f2bf(xh);
}

// =========================== per-step LSTM cell (WMMA) ===========================
// wave -> (btile, htile); 4 gate accumulators; 64 v_wmma per wave.
// gates = x_hat@W_ih^T + h@W_hh^T + bsum; pointwise sigmoid/tanh fused.
// hc ping-pong (hc_in -> hc_out) avoids read/write race; c kept in f32.
__global__ void __launch_bounds__(128) k_lstm(const u16* __restrict__ xhbf,
                                              const u16* __restrict__ hc_in,
                                              u16* __restrict__ hc_out,
                                              const u16* __restrict__ Wihbf,
                                              const u16* __restrict__ Whhbf,
                                              const float* __restrict__ bsum,
                                              float* __restrict__ c_state,
                                              float* __restrict__ out_enc, int t) {
    const int wave  = blockIdx.x * 4 + (threadIdx.x >> 5);
    const int btile = wave >> 4;     // 0..31
    const int htile = wave & 15;     // 0..15
    const int lane  = threadIdx.x & 31;
    const int lhalf = (lane & 16) ? 1 : 0;
    const int l15   = lane & 15;

    v8f z = {0.f, 0.f, 0.f, 0.f, 0.f, 0.f, 0.f, 0.f};
    v8f acc[4];
#pragma unroll
    for (int g = 0; g < 4; ++g) acc[g] = z;

    const int brow = btile * 16 + l15;
#pragma unroll
    for (int p = 0; p < 2; ++p) {
        const u16* Arow = p ? (hc_in + brow * (2 * HH)) : (xhbf + brow * KK);
        const u16* W    = p ? Whhbf : Wihbf;
#pragma unroll
        for (int c = 0; c < 8; ++c) {
            const int ct = c * 32;
            const u16* ap = Arow + ct + lhalf * 8;
            v16u afr = ld16(ap, ap + 16);
#pragma unroll
            for (int g = 0; g < 4; ++g) {
                const u16* bp = W + (g * HH + htile * 16 + l15) * KK + ct + lhalf * 8;
                v16u bfr = ld16(bp, bp + 16);
                acc[g] = wmma_bf16(afr, bfr, acc[g]);
            }
        }
    }

    const int hcol = htile * 16 + l15;
    const float bi = bsum[0 * HH + hcol], bff = bsum[1 * HH + hcol];
    const float bg = bsum[2 * HH + hcol], bo  = bsum[3 * HH + hcol];
#pragma unroll
    for (int r = 0; r < 8; ++r) {
        const int bidx = btile * 16 + r + lhalf * 8;
        float iv = fast_sigm(acc[0][r] + bi);
        float fv = fast_sigm(acc[1][r] + bff);
        float gv = fast_tanh(acc[2][r] + bg);
        float ov = fast_sigm(acc[3][r] + bo);
        float co = c_state[bidx * HH + hcol];
        float c2 = fv * co + iv * gv;
        float h2 = ov * fast_tanh(c2);
        c_state[bidx * HH + hcol] = c2;
        hc_out[bidx * (2 * HH) + hcol] = f2bf(h2);
        hc_out[bidx * (2 * HH) + HH + hcol] = f2bf(c2);
        out_enc[(size_t)bidx * (TM1 * HH) + t * HH + hcol] = h2;
    }
}

// =========================== host launcher ===========================
extern "C" void kernel_launch(void* const* d_in, const int* in_sizes, int n_in,
                              void* d_out, int out_size, void* d_ws, size_t ws_size,
                              hipStream_t stream) {
    (void)in_sizes; (void)n_in; (void)out_size; (void)ws_size;
    const float* input  = (const float*)d_in[0];
    const float* vew    = (const float*)d_in[1];
    /* d_in[2] = v_e_b: softmax shift-invariant, unused */
    const float* Wehs_w = (const float*)d_in[3];
    const float* Wehs_b = (const float*)d_in[4];
    const float* Ue_w   = (const float*)d_in[5];
    const float* Ue_b   = (const float*)d_in[6];
    const float* Wih    = (const float*)d_in[7];
    const float* Whh    = (const float*)d_in[8];
    const float* bih    = (const float*)d_in[9];
    const float* bhh    = (const float*)d_in[10];
    float* out = (float*)d_out;
    char*  ws  = (char*)d_ws;

    size_t off = 0;
    auto alloc = [&](size_t bytes) -> char* {
        char* p = ws + off;
        off += (bytes + 255) & ~(size_t)255;
        return p;
    };
    u16*   u_t    = (u16*)  alloc((size_t)BB * SP * KK * 2);   // 64 MB, L2-resident
    u16*   hc0    = (u16*)  alloc((size_t)BB * 2 * HH * 2);
    u16*   hc1    = (u16*)  alloc((size_t)BB * 2 * HH * 2);
    float* cst    = (float*)alloc((size_t)BB * HH * 4);
    u16*   xhbf   = (u16*)  alloc((size_t)BB * KK * 2);
    u16*   Wehsbf = (u16*)  alloc((size_t)TM1 * 2 * HH * 2);
    u16*   Uebf   = (u16*)  alloc((size_t)SP * SP * 2);
    u16*   Wihbf  = (u16*)  alloc((size_t)4 * HH * KK * 2);
    u16*   Whhbf  = (u16*)  alloc((size_t)4 * HH * KK * 2);
    float* bias2  = (float*)alloc((size_t)SP * 4);
    float* bsum   = (float*)alloc((size_t)4 * HH * 4);

    // setup
    k_init<<<BB * HH / 256, 256, 0, stream>>>(cst, hc0);
    k_cvt<<<(TM1 * 2 * HH + 255) / 256, 256, 0, stream>>>(Wehsbf, Wehs_w, TM1 * 2 * HH);
    k_cvt<<<(4 * HH * KK + 255) / 256, 256, 0, stream>>>(Wihbf, Wih, 4 * HH * KK);
    k_cvt<<<(4 * HH * KK + 255) / 256, 256, 0, stream>>>(Whhbf, Whh, 4 * HH * KK);
    k_build_ue<<<SP * SP / 256, 256, 0, stream>>>(Uebf, Ue_w);
    k_build_bias<<<4, 256, 0, stream>>>(bias2, bsum, Wehs_b, Ue_b, bih, bhh);

    // step-invariant attention pre-GEMM (WMMA)
    k_ugemm<<<BB * 16, 128, 0, stream>>>(u_t, Uebf, input);

    // 255 sequential steps: attention -> LSTM cell
    float* out_enc = out + OUT_ENC_OFF;
    for (int t = 0; t < TM1; ++t) {
        u16* hin  = (t & 1) ? hc1 : hc0;
        u16* hout = (t & 1) ? hc0 : hc1;
        k_attn<<<BB, 256, 0, stream>>>(u_t, hin, Wehsbf, bias2, vew, input,
                                       out, xhbf, t);
        k_lstm<<<128, 128, 0, stream>>>(xhbf, hin, hout, Wihbf, Whhbf, bsum,
                                        cst, out_enc, t);
    }
}
